// SpGATv2_46901042872920
// MI455X (gfx1250) — matrix operand order repaired and checked by reading, not compiled, not run on hardware
//
#include <hip/hip_runtime.h>
#include <hip/hip_bf16.h>
#include <stdint.h>

#define NNODES 50000
#define NFEAT  512
#define NHID   64
#define NHEADS 8
#define NCLASS 40
#define LRELU_A 0.2f

typedef __attribute__((ext_vector_type(16))) __bf16       bf16x16;
typedef __attribute__((ext_vector_type(8)))  float        f32x8;
typedef __attribute__((ext_vector_type(4)))  unsigned int u32x4;

union Frag { bf16x16 v; u32x4 q[2]; };

__device__ __forceinline__ unsigned short f2bf(float f) {
  unsigned int u = __float_as_uint(f);
  u += 0x7FFFu + ((u >> 16) & 1u);           // RNE
  return (unsigned short)(u >> 16);
}
__device__ __forceinline__ float elu1(float x)  { return x > 0.f ? x : (__expf(x) - 1.f); }
__device__ __forceinline__ float lrelu(float x) { return x > 0.f ? x : LRELU_A * x; }

// ---------------------------------------------------------------- converts --
__global__ void k_f32_to_bf16(const float* __restrict__ in,
                              unsigned short* __restrict__ out, int n) {
  int i = blockIdx.x * blockDim.x + threadIdx.x;
  if (i < n) out[i] = f2bf(in[i]);
}

// ---------------------------------------------------------- weight packers --
// Packed B layout per 32(K)x16(N) tile: lane l -> col = l&15,
// kbase = (l>>4)*16; dword d (0..7) holds bf16 pair (kbase+2d, kbase+2d+1).
// out flat index == gid for each packer below.

// W: [8,512,64] -> B [512 K][512 cols] with col c = head*64+hid.  16 kt x 32 nt.
__global__ void k_pack_w1(const float* __restrict__ W, unsigned int* __restrict__ out) {
  int gid = blockIdx.x * blockDim.x + threadIdx.x;
  if (gid >= 16 * 32 * 32 * 8) return;
  int d = gid & 7, lane = (gid >> 3) & 31, nt = (gid >> 8) & 31, kt = gid >> 13;
  int col = nt * 16 + (lane & 15);
  int k0  = kt * 32 + ((lane >> 4) << 4) + d * 2;
  int head = col >> 6, hid = col & 63;
  float v0 = W[(head * 512 + k0)     * 64 + hid];
  float v1 = W[(head * 512 + k0 + 1) * 64 + hid];
  out[gid] = (unsigned int)f2bf(v0) | ((unsigned int)f2bf(v1) << 16);
}

// W_pair: [8,128,64]; half t in {0:src,1:dst}.  2 kt x 4 nt per (h,t).
__global__ void k_pack_wpair(const float* __restrict__ Wp, unsigned int* __restrict__ out) {
  int gid = blockIdx.x * blockDim.x + threadIdx.x;
  if (gid >= 8 * 2 * 2 * 4 * 32 * 8) return;
  int d = gid & 7, lane = (gid >> 3) & 31, nt = (gid >> 8) & 3;
  int kt = (gid >> 10) & 1, t = (gid >> 11) & 1, h = gid >> 12;
  int col = nt * 16 + (lane & 15);
  int k0  = kt * 32 + ((lane >> 4) << 4) + d * 2;
  float v0 = Wp[(h * 128 + t * 64 + k0)     * 64 + col];
  float v1 = Wp[(h * 128 + t * 64 + k0 + 1) * 64 + col];
  out[gid] = (unsigned int)f2bf(v0) | ((unsigned int)f2bf(v1) << 16);
}

// W_out: [512,40] -> pad cols to 48.  16 kt x 3 nt.
__global__ void k_pack_wout(const float* __restrict__ Wo, unsigned int* __restrict__ out) {
  int gid = blockIdx.x * blockDim.x + threadIdx.x;
  if (gid >= 16 * 3 * 32 * 8) return;
  int d = gid & 7, lane = (gid >> 3) & 31, rest = gid >> 8;
  int nt = rest % 3, kt = rest / 3;
  int col = nt * 16 + (lane & 15);
  int k0  = kt * 32 + ((lane >> 4) << 4) + d * 2;
  float v0 = (col < 40) ? Wo[k0       * 40 + col] : 0.f;
  float v1 = (col < 40) ? Wo[(k0 + 1) * 40 + col] : 0.f;
  out[gid] = (unsigned int)f2bf(v0) | ((unsigned int)f2bf(v1) << 16);
}

// W_pair_out: [80,40]; halves t, K padded 40->64 (2 kt), cols padded to 48 (3 nt).
__global__ void k_pack_wpo(const float* __restrict__ Wpo, unsigned int* __restrict__ out) {
  int gid = blockIdx.x * blockDim.x + threadIdx.x;
  if (gid >= 2 * 2 * 3 * 32 * 8) return;
  int d = gid & 7, lane = (gid >> 3) & 31, rest = gid >> 8;
  int nt = rest % 3, q = rest / 3, kt = q & 1, t = q >> 1;
  int col = nt * 16 + (lane & 15);
  int k0  = kt * 32 + ((lane >> 4) << 4) + d * 2;
  float v0 = (col < 40 && k0     < 40) ? Wpo[(t * 40 + k0)     * 40 + col] : 0.f;
  float v1 = (col < 40 && k0 + 1 < 40) ? Wpo[(t * 40 + k0 + 1) * 40 + col] : 0.f;
  out[gid] = (unsigned int)f2bf(v0) | ((unsigned int)f2bf(v1) << 16);
}

// -------------------------------------------------------------- WMMA GEMMs --
// A fragment (16x32 bf16): lane l -> row = base+(l&15), dw0-3 = K kb..kb+7,
// dw4-7 = K kb+8..kb+15 with kb = k0 + 16*(l>>4).  Both halves are 16B loads.
__device__ __forceinline__ Frag load_a(const unsigned short* __restrict__ A,
                                       int row, int lda, int k0, int lane) {
  Frag a;
  const unsigned short* p = A + (size_t)row * lda + k0 + ((lane >> 4) << 4);
  a.q[0] = *(const u32x4*)p;
  a.q[1] = *(const u32x4*)(p + 8);
  return a;
}
__device__ __forceinline__ Frag load_b(const unsigned int* __restrict__ Bpk,
                                       int tileIdx, int lane) {
  Frag b;
  const u32x4* p = (const u32x4*)(Bpk + ((size_t)tileIdx * 32 + lane) * 8);
  b.q[0] = p[0]; b.q[1] = p[1];
  return b;
}
#define WMMA_BF16(A, B, C) __builtin_amdgcn_wmma_f32_16x16x32_bf16( \
    false, (A), false, (B), (short)0, (C), false, false)

// h[N,512] = xbf[N,512] @ W1pk.  grid = 3125 blocks x 256 thr; wave = colgroup.
// Software-pipelined: loads for kt+1 in flight while WMMAs consume kt.
__global__ void k_gemm1(const unsigned short* __restrict__ Xbf,
                        const unsigned int* __restrict__ W1pk,
                        float* __restrict__ H) {
  int wave = threadIdx.x >> 5, lane = threadIdx.x & 31;
  int tileM = blockIdx.x, cg = wave;
  int rowA = tileM * 16 + (lane & 15);
  f32x8 acc[4] = {};
  Frag a_cur = load_a(Xbf, rowA, 512, 0, lane);
  Frag b_cur[4];
#pragma unroll
  for (int j = 0; j < 4; ++j) b_cur[j] = load_b(W1pk, cg * 4 + j, lane);
#pragma unroll 3
  for (int kt = 0; kt < 15; ++kt) {
    Frag a_nxt = load_a(Xbf, rowA, 512, (kt + 1) * 32, lane);
    Frag b_nxt[4];
#pragma unroll
    for (int j = 0; j < 4; ++j)
      b_nxt[j] = load_b(W1pk, (kt + 1) * 32 + cg * 4 + j, lane);
#pragma unroll
    for (int j = 0; j < 4; ++j) acc[j] = WMMA_BF16(a_cur.v, b_cur[j].v, acc[j]);
    a_cur = a_nxt;
#pragma unroll
    for (int j = 0; j < 4; ++j) b_cur[j] = b_nxt[j];
  }
#pragma unroll
  for (int j = 0; j < 4; ++j) acc[j] = WMMA_BF16(a_cur.v, b_cur[j].v, acc[j]);
  int rowBase = tileM * 16 + ((lane >> 4) << 3);
  int colBase = cg * 64 + (lane & 15);
#pragma unroll
  for (int j = 0; j < 4; ++j)
#pragma unroll
    for (int v = 0; v < 8; ++v)
      H[(size_t)(rowBase + v) * 512 + colBase + j * 16] = acc[j][v];
}

// hs/hd[N,512]: per head, h_bf[:,64h:64h+64] @ Wpair halves.  wave = head.
__global__ void k_gemm2(const unsigned short* __restrict__ Hbf,
                        const unsigned int* __restrict__ Wppk,
                        float* __restrict__ HS, float* __restrict__ HD) {
  int head = threadIdx.x >> 5, lane = threadIdx.x & 31;
  int tileM = blockIdx.x;
  int rowA = tileM * 16 + (lane & 15);
  f32x8 cs[4] = {}, cd[4] = {};
#pragma unroll
  for (int kt = 0; kt < 2; ++kt) {
    Frag a = load_a(Hbf + head * 64, rowA, 512, kt * 32, lane);
#pragma unroll
    for (int j = 0; j < 4; ++j) {
      Frag bs = load_b(Wppk, ((head * 2 + 0) * 2 + kt) * 4 + j, lane);
      Frag bd = load_b(Wppk, ((head * 2 + 1) * 2 + kt) * 4 + j, lane);
      cs[j] = WMMA_BF16(a.v, bs.v, cs[j]);
      cd[j] = WMMA_BF16(a.v, bd.v, cd[j]);
    }
  }
  int rowBase = tileM * 16 + ((lane >> 4) << 3);
  int colBase = head * 64 + (lane & 15);
#pragma unroll
  for (int j = 0; j < 4; ++j)
#pragma unroll
    for (int v = 0; v < 8; ++v) {
      size_t idx = (size_t)(rowBase + v) * 512 + colBase + j * 16;
      HS[idx] = cs[j][v];
      HD[idx] = cd[j][v];
    }
}

// g[N,40] (+ gbf[N,64] padded) = x2bf @ Wopk.  one wave per row-tile; pipelined.
__global__ void k_gemm3(const unsigned short* __restrict__ X2bf,
                        const unsigned int* __restrict__ Wopk,
                        float* __restrict__ G, unsigned short* __restrict__ Gbf) {
  int wave = threadIdx.x >> 5, lane = threadIdx.x & 31;
  int gw = blockIdx.x * 8 + wave;
  if (gw >= NNODES / 16) return;
  int rowA = gw * 16 + (lane & 15);
  f32x8 acc[3] = {};
  Frag a_cur = load_a(X2bf, rowA, 512, 0, lane);
  Frag b_cur[3];
#pragma unroll
  for (int j = 0; j < 3; ++j) b_cur[j] = load_b(Wopk, j, lane);
#pragma unroll 3
  for (int kt = 0; kt < 15; ++kt) {
    Frag a_nxt = load_a(X2bf, rowA, 512, (kt + 1) * 32, lane);
    Frag b_nxt[3];
#pragma unroll
    for (int j = 0; j < 3; ++j) b_nxt[j] = load_b(Wopk, (kt + 1) * 3 + j, lane);
#pragma unroll
    for (int j = 0; j < 3; ++j) acc[j] = WMMA_BF16(a_cur.v, b_cur[j].v, acc[j]);
    a_cur = a_nxt;
#pragma unroll
    for (int j = 0; j < 3; ++j) b_cur[j] = b_nxt[j];
  }
#pragma unroll
  for (int j = 0; j < 3; ++j) acc[j] = WMMA_BF16(a_cur.v, b_cur[j].v, acc[j]);
  int rowBase = gw * 16 + ((lane >> 4) << 3);
  int colIn = lane & 15;
#pragma unroll
  for (int j = 0; j < 3; ++j)
#pragma unroll
    for (int v = 0; v < 8; ++v) {
      int col = j * 16 + colIn;
      if (col < NCLASS) {
        int row = rowBase + v;
        G[(size_t)row * NCLASS + col] = acc[j][v];
        Gbf[(size_t)row * 64 + col] = f2bf(acc[j][v]);
      }
    }
}

// gs/gd[N,40] = gbf[N,64] @ Wpopk halves (K padded to 64).
__global__ void k_gemm4(const unsigned short* __restrict__ Gbf,
                        const unsigned int* __restrict__ Wpopk,
                        float* __restrict__ GS, float* __restrict__ GD) {
  int wave = threadIdx.x >> 5, lane = threadIdx.x & 31;
  int gw = blockIdx.x * 8 + wave;
  if (gw >= NNODES / 16) return;
  int rowA = gw * 16 + (lane & 15);
  f32x8 cs[3] = {}, cd[3] = {};
#pragma unroll
  for (int kt = 0; kt < 2; ++kt) {
    Frag a = load_a(Gbf, rowA, 64, kt * 32, lane);
#pragma unroll
    for (int j = 0; j < 3; ++j) {
      Frag bs = load_b(Wpopk, (0 * 2 + kt) * 3 + j, lane);
      Frag bd = load_b(Wpopk, (1 * 2 + kt) * 3 + j, lane);
      cs[j] = WMMA_BF16(a.v, bs.v, cs[j]);
      cd[j] = WMMA_BF16(a.v, bd.v, cd[j]);
    }
  }
  int rowBase = gw * 16 + ((lane >> 4) << 3);
  int colIn = lane & 15;
#pragma unroll
  for (int j = 0; j < 3; ++j)
#pragma unroll
    for (int v = 0; v < 8; ++v) {
      int col = j * 16 + colIn;
      if (col < NCLASS) {
        size_t idx = (size_t)(rowBase + v) * NCLASS + col;
        GS[idx] = cs[j][v];
        GD[idx] = cd[j][v];
      }
    }
}

// ------------------------------------------------------------- edge phases --
// wave per (edge, head); lane owns 2 contiguous dims of 64.
__global__ void k_edge1(const int* __restrict__ src, const int* __restrict__ dst,
                        const float* __restrict__ HS, const float* __restrict__ HD,
                        const float* __restrict__ H, const float* __restrict__ A,
                        float* __restrict__ HP, float* __restrict__ RS, long long EH) {
  long long w = ((long long)blockIdx.x * blockDim.x + threadIdx.x) >> 5;
  if (w >= EH) return;
  int lane = threadIdx.x & 31;
  int e = (int)(w >> 3), head = (int)(w & 7);
  int s = src[e], t = dst[e];
  int d0 = head * 64 + lane * 2;
  float2 vs = *(const float2*)(HS + (size_t)s * 512 + d0);
  float2 vd = *(const float2*)(HD + (size_t)t * 512 + d0);
  float2 va = *(const float2*)(A + head * 64 + lane * 2);
  float p = va.x * lrelu(vs.x + vd.x) + va.y * lrelu(vs.y + vd.y);
#pragma unroll
  for (int off = 16; off; off >>= 1) p += __shfl_xor(p, off, 32);
  float att = __expf(p);
  float2 vh = *(const float2*)(H + (size_t)t * 512 + d0);
  atomicAdd(HP + (size_t)s * 512 + d0,     att * vh.x);
  atomicAdd(HP + (size_t)s * 512 + d0 + 1, att * vh.y);
  if (lane == 0) atomicAdd(RS + (size_t)s * 8 + head, att);
}

// wave per edge; 40 dims -> lane covers d=lane and (lane<8) d=lane+32.
__global__ void k_edge2(const int* __restrict__ src, const int* __restrict__ dst,
                        const float* __restrict__ GS, const float* __restrict__ GD,
                        const float* __restrict__ G, const float* __restrict__ AO,
                        float* __restrict__ HP2, float* __restrict__ RS2, long long E) {
  long long w = ((long long)blockIdx.x * blockDim.x + threadIdx.x) >> 5;
  if (w >= E) return;
  int lane = threadIdx.x & 31;
  int s = src[w], t = dst[w];
  float p = AO[lane] * lrelu(GS[(size_t)s * 40 + lane] + GD[(size_t)t * 40 + lane]);
  if (lane < 8) {
    int d = lane + 32;
    p += AO[d] * lrelu(GS[(size_t)s * 40 + d] + GD[(size_t)t * 40 + d]);
  }
#pragma unroll
  for (int off = 16; off; off >>= 1) p += __shfl_xor(p, off, 32);
  float att = __expf(p);
  atomicAdd(HP2 + (size_t)s * 40 + lane, att * G[(size_t)t * 40 + lane]);
  if (lane < 8) {
    int d = lane + 32;
    atomicAdd(HP2 + (size_t)s * 40 + d, att * G[(size_t)t * 40 + d]);
  }
  if (lane == 0) atomicAdd(RS2 + s, att);
}

// ---------------------------------------------------------------- finalize --
__global__ void k_final1(const float* __restrict__ HP, const float* __restrict__ RS,
                         float* __restrict__ X2f, unsigned short* __restrict__ X2bf) {
  int i = blockIdx.x * blockDim.x + threadIdx.x;
  if (i >= NNODES * 512) return;
  int n = i >> 9, head = (i >> 6) & 7;
  float v = HP[i] / RS[n * 8 + head];
  float y = elu1(elu1(v));          // per-head elu, then elu of concat
  X2f[i] = y;
  X2bf[i] = f2bf(y);
}

__global__ void k_final2(const float* __restrict__ HP2, const float* __restrict__ RS2,
                         float* __restrict__ OUT) {
  int i = blockIdx.x * blockDim.x + threadIdx.x;
  if (i >= NNODES * NCLASS) return;
  int n = i / NCLASS;
  OUT[i] = elu1(HP2[i] / RS2[n]);
}

// ------------------------------------------------------------------ launch --
extern "C" void kernel_launch(void* const* d_in, const int* in_sizes, int n_in,
                              void* d_out, int out_size, void* d_ws, size_t ws_size,
                              hipStream_t stream) {
  const float* x   = (const float*)d_in[0];
  const int*   edg = (const int*)d_in[1];
  const float* W   = (const float*)d_in[2];
  const float* Wp  = (const float*)d_in[3];
  const float* av  = (const float*)d_in[4];
  const float* Wo  = (const float*)d_in[5];
  const float* Wpo = (const float*)d_in[6];
  const float* ao  = (const float*)d_in[7];
  float* out = (float*)d_out;

  const long long E = in_sizes[1] / 2;
  const int* src = edg;
  const int* dst = edg + E;

  char* ws = (char*)d_ws;
  float*          h      = (float*)(ws);
  unsigned short* bfbuf  = (unsigned short*)(ws + 102400000LL); // xbf -> hbf -> x2bf
  float*          hs     = (float*)(ws + 153600000LL);          // -> x2f
  char*           l2     = ws + 256000000LL;                    // hd -> layer-2 pool
  float*          hd     = (float*)l2;
  float*          hp     = (float*)(ws + 358400000LL);
  float*          rowsum = (float*)(ws + 460800000LL);
  unsigned int*   W1pk   = (unsigned int*)(ws + 462400000LL);
  unsigned int*   Wppk   = (unsigned int*)(ws + 462924288LL);
  unsigned int*   Wopk   = (unsigned int*)(ws + 463055360LL);
  unsigned int*   Wpopk  = (unsigned int*)(ws + 463104512LL);

  float*          g    = (float*)(l2 + 0);
  float*          gs   = (float*)(l2 + 8000000);
  float*          gd   = (float*)(l2 + 16000000);
  float*          hp2  = (float*)(l2 + 24000000);
  float*          rs2  = (float*)(l2 + 32000000);
  unsigned short* gbf  = (unsigned short*)(l2 + 32200000);

  // --- pack weights + convert x ---
  k_f32_to_bf16<<<(NNODES * 512 + 255) / 256, 256, 0, stream>>>(x, bfbuf, NNODES * 512);
  k_pack_w1   <<<(16 * 32 * 256 + 255) / 256, 256, 0, stream>>>(W, W1pk);
  k_pack_wpair<<<(8 * 2 * 2 * 4 * 256 + 255) / 256, 256, 0, stream>>>(Wp, Wppk);
  k_pack_wout <<<(16 * 3 * 256 + 255) / 256, 256, 0, stream>>>(Wo, Wopk);
  k_pack_wpo  <<<(2 * 2 * 3 * 256 + 255) / 256, 256, 0, stream>>>(Wpo, Wpopk);

  // --- layer 1 ---
  k_gemm1<<<NNODES / 16, 256, 0, stream>>>(bfbuf, W1pk, h);
  k_f32_to_bf16<<<(NNODES * 512 + 255) / 256, 256, 0, stream>>>(h, bfbuf, NNODES * 512);
  k_gemm2<<<NNODES / 16, 256, 0, stream>>>(bfbuf, Wppk, hs, hd);

  hipMemsetAsync(hp, 0, (size_t)NNODES * 512 * 4, stream);
  hipMemsetAsync(rowsum, 0, (size_t)NNODES * 8 * 4, stream);

  long long EH = E * NHEADS;
  unsigned int blocks1 = (unsigned int)((EH + 7) / 8);
  k_edge1<<<blocks1, 256, 0, stream>>>(src, dst, hs, hd, h, av, hp, rowsum, EH);

  // x2 -> reuse hs (f32) and bfbuf (bf16)
  k_final1<<<(NNODES * 512 + 255) / 256, 256, 0, stream>>>(hp, rowsum, hs, bfbuf);

  // --- layer 2 ---
  hipMemsetAsync(gbf, 0, (size_t)NNODES * 64 * 2, stream);
  int gw_blocks = (NNODES / 16 + 7) / 8;
  k_gemm3<<<gw_blocks, 256, 0, stream>>>(bfbuf, Wopk, g, gbf);
  k_gemm4<<<gw_blocks, 256, 0, stream>>>(gbf, Wpopk, gs, gd);

  hipMemsetAsync(hp2, 0, (size_t)NNODES * 40 * 4, stream);
  hipMemsetAsync(rs2, 0, (size_t)NNODES * 4, stream);

  unsigned int blocks2 = (unsigned int)((E + 7) / 8);
  k_edge2<<<blocks2, 256, 0, stream>>>(src, dst, gs, gd, g, ao, hp2, rs2, E);

  k_final2<<<(NNODES * NCLASS + 255) / 256, 256, 0, stream>>>(hp2, rs2, out);
}
